// GmuAttention_43284680409313
// MI455X (gfx1250) — compile-verified, matrix operand order checked by reading, not run on hardware
//
#include <hip/hip_runtime.h>
#include <hip/hip_bf16.h>
#include <stdint.h>

// ---------------------------------------------------------------------------
// CDNA5 (gfx1250) fused GMU-attention pipeline.
// All GEMMs run on v_wmma_f32_16x16x32_bf16 (wave32, 16x16 tiles, f32 accum).
// ---------------------------------------------------------------------------

typedef __attribute__((ext_vector_type(16))) __bf16 v16bf;
typedef __attribute__((ext_vector_type(8)))  float  v8f;

#define TM 128
#define TN 128
#define TK 32
#define KSTRIDE 40   // padded LDS row stride in elements (80B, multiple of 16B)

struct alignas(16) U4 { uint32_t x, y, z, w; };
union B16x8 { U4 v; __bf16 h[8]; };
union F32x4 { U4 v; float f[4]; };
union B16x4 { uint64_t q; __bf16 h[4]; };

__device__ __forceinline__ float silu_(float x) {
  return x * (1.f / (1.f + __expf(-x)));
}

// Load one 16x32 bf16 WMMA fragment from an LDS row (row-major, KSTRIDE pitch).
// Lane layout per CDNA5 ISA: lanes 0-15 hold K in [0,8)+[16,24),
// lanes 16-31 hold K in [8,16)+[24,32). Two ds_load_b128 per fragment.
__device__ __forceinline__ v16bf frag_ld(const __bf16* p, int ka) {
  v16bf f;
  __builtin_memcpy(&f, p + ka, 16);
  __builtin_memcpy(reinterpret_cast<char*>(&f) + 16, p + ka + 16, 16);
  return f;
}

// EP: 1 = bf16 store (acc+bias)
//     2 = bf16 store silu(acc+bias)
//     3 = bf16 store silu(acc+bias)*scale2+bias2, plus optional lambda-blend
//         with auxMem into blendOut (the q / memory path)
//     4 = f32 store (acc+bias) scattered back to (L, N, E) layout
template<bool ATRANS, bool BBF16, int EP>
__global__ __launch_bounds__(256)
void gemm_wmma(const void* __restrict__ Ap, long lda, long sA,
               const void* __restrict__ Bp, long ldb, long sB,
               void* __restrict__ Dp, long ldd, long sD,
               int M, int N, int K,
               const float* __restrict__ bias,
               const float* __restrict__ scale2,
               const float* __restrict__ bias2,
               const float* __restrict__ auxMem,
               __bf16* __restrict__ blendOut,
               int Lrows, int NbOut)
{
  __shared__ __attribute__((aligned(16))) __bf16 As[TM * KSTRIDE];
  __shared__ __attribute__((aligned(16))) __bf16 Bs[TN * KSTRIDE];

  const int tid   = threadIdx.x;
  const int lane  = tid & 31;
  const int wave  = tid >> 5;        // 8 waves (wave32)
  const int waveM = wave & 3;        // 4 waves along M -> 32 rows each
  const int waveN = wave >> 2;       // 2 waves along N -> 64 cols each
  const long rowBase = (long)blockIdx.y * TM;
  const long colBase = (long)blockIdx.x * TN;
  const int  zb = blockIdx.z;

  const __bf16* Ab = (const __bf16*)Ap + sA * zb;
  const __bf16* Bb = (const __bf16*)Bp + sB * zb;
  const float*  Bf = (const float*) Bp + sB * zb;

  // ---- per-thread staging descriptors: fixed 16B chunks, pointer-stepped ----
  // A tile: 128x32 bf16 = 512 chunks of 8 bf16, 2 per thread.
  const char* aPtr[2];
  int aLds[2];
  if (!ATRANS) {
#pragma unroll
    for (int i = 0; i < 2; ++i) {
      const int ch = tid + i * 256;
      const int r = ch >> 2, kc = (ch & 3) << 3;     // 4 chunks per row along K
      aPtr[i] = (const char*)(Ab + (rowBase + r) * lda + kc);
      aLds[i] = r * KSTRIDE + kc;
    }
  } else {
#pragma unroll
    for (int i = 0; i < 2; ++i) {
      const int ch = tid + i * 256;
      const int kk = ch >> 4, mc = (ch & 15) << 3;   // 16 chunks per K-row along M
      aPtr[i] = (const char*)(Ab + (long)kk * lda + rowBase + mc);
      aLds[i] = mc * KSTRIDE + kk;
    }
  }
  const long aStep = ATRANS ? (long)TK * lda * 2 : (long)TK * 2;

  // B tile: 32xK x 128 cols -> staged transposed as Bs[col][k].
  const char* bPtr[4];
  int bLds[4];
  if (BBF16) {
#pragma unroll
    for (int i = 0; i < 2; ++i) {
      const int ch = tid + i * 256;                  // 512 chunks of 8 bf16
      const int kk = ch >> 4, cc = (ch & 15) << 3;
      bPtr[i] = (const char*)(Bb + (long)kk * ldb + colBase + cc);
      bLds[i] = cc * KSTRIDE + kk;
    }
  } else {
#pragma unroll
    for (int i = 0; i < 4; ++i) {
      const int ch = tid + i * 256;                  // 1024 chunks of 4 f32
      const int kk = ch >> 5, cc = (ch & 31) << 2;
      bPtr[i] = (const char*)(Bf + (long)kk * ldb + colBase + cc);
      bLds[i] = cc * KSTRIDE + kk;
    }
  }
  const long bStep = (long)TK * ldb * (BBF16 ? 2 : 4);

  v8f acc[2][4];
#pragma unroll
  for (int i = 0; i < 2; ++i)
#pragma unroll
    for (int j = 0; j < 4; ++j)
#pragma unroll
      for (int e = 0; e < 8; ++e) acc[i][j][e] = 0.f;

  const int fr = lane & 15;
  const int ka = (lane & 16) ? 8 : 0;

  for (int k0 = 0; k0 < K; k0 += TK) {
    // ---- stage A (global_load_b128 -> ds_store_b128 / b16-scatter) ----
    if (!ATRANS) {
#pragma unroll
      for (int i = 0; i < 2; ++i) {
        const U4 v = *reinterpret_cast<const U4*>(aPtr[i]);
        *reinterpret_cast<U4*>(&As[aLds[i]]) = v;
        aPtr[i] += aStep;
      }
    } else {
#pragma unroll
      for (int i = 0; i < 2; ++i) {
        B16x8 u; u.v = *reinterpret_cast<const U4*>(aPtr[i]);
#pragma unroll
        for (int j = 0; j < 8; ++j) As[aLds[i] + j * KSTRIDE] = u.h[j];
        aPtr[i] += aStep;
      }
    }
    // ---- stage B transposed ----
    if (BBF16) {
#pragma unroll
      for (int i = 0; i < 2; ++i) {
        B16x8 u; u.v = *reinterpret_cast<const U4*>(bPtr[i]);
#pragma unroll
        for (int j = 0; j < 8; ++j) Bs[bLds[i] + j * KSTRIDE] = u.h[j];
        bPtr[i] += bStep;
      }
    } else {
#pragma unroll
      for (int i = 0; i < 4; ++i) {
        F32x4 u; u.v = *reinterpret_cast<const U4*>(bPtr[i]);
#pragma unroll
        for (int j = 0; j < 4; ++j) Bs[bLds[i] + j * KSTRIDE] = (__bf16)u.f[j];
        bPtr[i] += bStep;
      }
    }

    if (k0 + TK < K) {   // pointers already advanced -> touch next slab
      __builtin_prefetch(aPtr[0], 0, 1);
      __builtin_prefetch(bPtr[0], 0, 1);
    }
    __syncthreads();

    v16bf afrag[2], bfrag[4];
#pragma unroll
    for (int im = 0; im < 2; ++im)
      afrag[im] = frag_ld(&As[(waveM * 32 + im * 16 + fr) * KSTRIDE], ka);
#pragma unroll
    for (int jn = 0; jn < 4; ++jn)
      bfrag[jn] = frag_ld(&Bs[(waveN * 64 + jn * 16 + fr) * KSTRIDE], ka);

#pragma unroll
    for (int im = 0; im < 2; ++im)
#pragma unroll
      for (int jn = 0; jn < 4; ++jn)
        acc[im][jn] = __builtin_amdgcn_wmma_f32_16x16x32_bf16(
            false, afrag[im], false, bfrag[jn], (short)0, acc[im][jn],
            false, false);

    __syncthreads();
  }

  // ---- epilogue ----
  // 16x16 f32 D layout: VGPR r, lanes 0-15 -> M=r, lanes 16-31 -> M=8+r; N=lane&15
  float*  Df = (float*) Dp + sD * zb;
  __bf16* Db = (__bf16*)Dp + sD * zb;
  const int em = (lane & 16) ? 8 : 0;
  const int en = lane & 15;
#pragma unroll
  for (int im = 0; im < 2; ++im) {
#pragma unroll
    for (int jn = 0; jn < 4; ++jn) {
#pragma unroll
      for (int r = 0; r < 8; ++r) {
        const long row = rowBase + waveM * 32 + im * 16 + em + r;
        const long col = colBase + waveN * 64 + jn * 16 + en;
        float v = acc[im][jn][r];
        if (bias) v += bias[col];
        if (EP == 1) {
          Db[row * ldd + col] = (__bf16)v;
        } else if (EP == 2) {
          Db[row * ldd + col] = (__bf16)silu_(v);
        } else if (EP == 3) {
          const float vq = silu_(v) * scale2[col] + bias2[col];
          Db[row * ldd + col] = (__bf16)vq;
          if (blendOut) {
            const long s = row % Lrows;           // row = n*L + s
            blendOut[row * ldd + col] =
                (__bf16)(0.001f * auxMem[s * ldd + col] + 0.999f * vq);
          }
        } else { // EP == 4 : scatter (n*L+l, col) -> out[l][n][col], f32
          const long l = row % Lrows, n = row / Lrows;
          Df[(l * NbOut + n) * ldd + col] = v;
        }
      }
    }
  }
}

// ---------------- LayerNorm kernels (one row per block, 256 threads) --------

__device__ __forceinline__ float block_sum(float v, float* red) {
#pragma unroll
  for (int o = 16; o > 0; o >>= 1) v += __shfl_xor(v, o, 32);  // wave32
  const int w = threadIdx.x >> 5;
  if ((threadIdx.x & 31) == 0) red[w] = v;
  __syncthreads();
  float t = 0.f;
#pragma unroll
  for (int i = 0; i < 8; ++i) t += red[i];
  __syncthreads();
  return t;
}

// src is (L, Nb, E) f32; dst is (Nb*L, E) bf16 row-major (row = n*L + l)
// NCH chunks of 4 f32 per thread; E == NCH * 1024
template<int NCH>
__global__ __launch_bounds__(256)
void ln_kernel(const float* __restrict__ src,
               const float* __restrict__ w, const float* __restrict__ b,
               __bf16* __restrict__ dst, int Lr, int Nb, int E)
{
  __shared__ float red[8];
  const long row = blockIdx.x;
  const int  n = (int)(row / Lr), l = (int)(row % Lr);
  const float* x = src + ((long)l * Nb + n) * E;

  float vals[NCH * 4];
  float s = 0.f;
#pragma unroll
  for (int i = 0; i < NCH; ++i) {
    const float4 v = *reinterpret_cast<const float4*>(x + (threadIdx.x + i * 256) * 4);
    vals[i * 4 + 0] = v.x; vals[i * 4 + 1] = v.y;
    vals[i * 4 + 2] = v.z; vals[i * 4 + 3] = v.w;
    s += v.x + v.y + v.z + v.w;
  }
  const float mean = block_sum(s, red) / (float)E;
  float vv = 0.f;
#pragma unroll
  for (int i = 0; i < NCH * 4; ++i) { const float d = vals[i] - mean; vv += d * d; }
  const float rstd = rsqrtf(block_sum(vv, red) / (float)E + 1e-5f);

  __bf16* d0 = dst + row * (long)E;
#pragma unroll
  for (int i = 0; i < NCH; ++i) {
    const int c = (threadIdx.x + i * 256) * 4;
    B16x4 o;
#pragma unroll
    for (int j = 0; j < 4; ++j)
      o.h[j] = (__bf16)((vals[i * 4 + j] - mean) * rstd * w[c + j] + b[c + j]);
    *reinterpret_cast<uint64_t*>(d0 + c) = o.q;
  }
}

// z = LN(out1)*u ; src/u/dst all (Nb*L, E1) bf16 row-major
// NCH chunks of 8 bf16 per thread; E1 == NCH * 2048
template<int NCH>
__global__ __launch_bounds__(256)
void ln_mul_kernel(const __bf16* __restrict__ src,
                   const float* __restrict__ w, const float* __restrict__ b,
                   const __bf16* __restrict__ ug,
                   __bf16* __restrict__ dst, int E1)
{
  __shared__ float red[8];
  const long base = (long)blockIdx.x * E1;

  float vals[NCH * 8];
  float s = 0.f;
#pragma unroll
  for (int i = 0; i < NCH; ++i) {
    B16x8 u; u.v = *reinterpret_cast<const U4*>(src + base + (threadIdx.x + i * 256) * 8);
#pragma unroll
    for (int j = 0; j < 8; ++j) { vals[i * 8 + j] = (float)u.h[j]; s += vals[i * 8 + j]; }
  }
  const float mean = block_sum(s, red) / (float)E1;
  float vv = 0.f;
#pragma unroll
  for (int i = 0; i < NCH * 8; ++i) { const float d = vals[i] - mean; vv += d * d; }
  const float rstd = rsqrtf(block_sum(vv, red) / (float)E1 + 1e-5f);

#pragma unroll
  for (int i = 0; i < NCH; ++i) {
    const int c = (threadIdx.x + i * 256) * 8;
    B16x8 g; g.v = *reinterpret_cast<const U4*>(ug + base + c);
    B16x8 o;
#pragma unroll
    for (int j = 0; j < 8; ++j) {
      const float t = (vals[i * 8 + j] - mean) * rstd * w[c + j] + b[c + j];
      o.h[j] = (__bf16)(t * (float)g.h[j]);
    }
    *reinterpret_cast<U4*>(dst + base + c) = o.v;
  }
}

// ---------------------------------------------------------------------------

extern "C" void kernel_launch(void* const* d_in, const int* in_sizes, int n_in,
                              void* d_out, int out_size, void* d_ws, size_t ws_size,
                              hipStream_t stream) {
  (void)in_sizes; (void)n_in; (void)out_size; (void)ws_size;

  const float* query    = (const float*)d_in[0];
  const float* key_     = (const float*)d_in[1];
  /* d_in[2] (value) unused by the reference math */
  const float* norm_w   = (const float*)d_in[3];
  const float* norm_b   = (const float*)d_in[4];
  const float* u_w      = (const float*)d_in[5];
  const float* u_b      = (const float*)d_in[6];
  const float* qk_w     = (const float*)d_in[7];
  const float* qk_b     = (const float*)d_in[8];
  const float* q_weight = (const float*)d_in[9];
  const float* q_bias   = (const float*)d_in[10];
  const float* k_weight = (const float*)d_in[11];
  const float* k_bias   = (const float*)d_in[12];
  const float* memory   = (const float*)d_in[13];
  const float* mem_w    = (const float*)d_in[14];
  const float* mem_b    = (const float*)d_in[15];
  const float* after_w  = (const float*)d_in[16];
  const float* after_b  = (const float*)d_in[17];
  const float* o_w      = (const float*)d_in[18];
  const float* o_b      = (const float*)d_in[19];

  constexpr int E = 2048, E1 = 4096, E2 = 1024, L = 2048, Nb = 4;
  constexpr int R = Nb * L;   // 8192 flattened (n, l) rows

  size_t off = 0;
  auto alloc = [&](size_t nbytes) {
    void* p = (char*)d_ws + off;
    off += (nbytes + 255) & ~(size_t)255;
    return p;
  };
  __bf16* xn   = (__bf16*)alloc((size_t)R * E  * 2);  // LN(query)
  __bf16* yn   = (__bf16*)alloc((size_t)R * E  * 2);  // LN(key)
  __bf16* u_bf = (__bf16*)alloc((size_t)R * E1 * 2);  // silu gate
  __bf16* q_bf = (__bf16*)alloc((size_t)R * E2 * 2);
  __bf16* mbl  = (__bf16*)alloc((size_t)R * E2 * 2);  // lambda-blend(memory, q)
  __bf16* k_bf = (__bf16*)alloc((size_t)R * E2 * 2);
  __bf16* mv   = (__bf16*)alloc((size_t)R * E1 * 2);  // mem @ mem_w
  __bf16* o1   = (__bf16*)alloc((size_t)Nb * E2 * E1 * 2);
  __bf16* out1 = (__bf16*)alloc((size_t)R * E1 * 2);
  __bf16* z_bf = (__bf16*)alloc((size_t)R * E1 * 2);

  // 1) LayerNorms (also (L,N,E) -> (N*L,E) transpose, f32 -> bf16)
  ln_kernel<2><<<R, 256, 0, stream>>>(query, norm_w, norm_b, xn, L, Nb, E);
  ln_kernel<2><<<R, 256, 0, stream>>>(key_,  norm_w, norm_b, yn, L, Nb, E);

  // 2) u = silu(xn @ u_w + u_b)                       (8192 x 4096 x 2048)
  gemm_wmma<false, false, 2><<<dim3(E1 / TN, R / TM, 1), dim3(256), 0, stream>>>(
      xn, E, 0, u_w, E1, 0, u_bf, E1, 0, R, E1, E,
      u_b, nullptr, nullptr, nullptr, nullptr, L, Nb);

  // 3) q = silu(xn @ qk_w + qk_b)*q_w + q_b ; fused lambda-blend with memory
  gemm_wmma<false, false, 3><<<dim3(E2 / TN, R / TM, 1), dim3(256), 0, stream>>>(
      xn, E, 0, qk_w, E2, 0, q_bf, E2, 0, R, E2, E,
      qk_b, q_weight, q_bias, memory, mbl, L, Nb);

  // 4) k = silu(yn @ qk_w + qk_b)*k_w + k_b
  gemm_wmma<false, false, 3><<<dim3(E2 / TN, R / TM, 1), dim3(256), 0, stream>>>(
      yn, E, 0, qk_w, E2, 0, k_bf, E2, 0, R, E2, E,
      qk_b, k_weight, k_bias, nullptr, nullptr, L, Nb);

  // 5) mv = mbl @ mem_w + mem_b                       (8192 x 4096 x 1024)
  gemm_wmma<false, false, 1><<<dim3(E1 / TN, R / TM, 1), dim3(256), 0, stream>>>(
      mbl, E2, 0, mem_w, E1, 0, mv, E1, 0, R, E1, E2,
      mem_b, nullptr, nullptr, nullptr, nullptr, L, Nb);

  // 6) o1[n] = k[n]^T @ mv[n]    (batched, A transposed; 1024 x 4096 x 2048)
  gemm_wmma<true, true, 1><<<dim3(E1 / TN, E2 / TM, Nb), dim3(256), 0, stream>>>(
      k_bf, E2, (long)L * E2, mv, E1, (long)L * E1, o1, E1, (long)E2 * E1,
      E2, E1, L, nullptr, nullptr, nullptr, nullptr, nullptr, L, Nb);

  // 7) out1[n] = q[n] @ o1[n]    (batched; 2048 x 4096 x 1024)
  gemm_wmma<false, true, 1><<<dim3(E1 / TN, L / TM, Nb), dim3(256), 0, stream>>>(
      q_bf, E2, (long)L * E2, o1, E1, (long)E2 * E1, out1, E1, (long)L * E1,
      L, E1, E2, nullptr, nullptr, nullptr, nullptr, nullptr, L, Nb);

  // 8) z = LN(out1; after_w, after_b) * u
  ln_mul_kernel<2><<<R, 256, 0, stream>>>(out1, after_w, after_b, u_bf, z_bf, E1);

  // 9) out = z @ o_w + o_b, scattered back to (L, N, E) f32
  gemm_wmma<false, false, 4><<<dim3(E / TN, R / TM, 1), dim3(256), 0, stream>>>(
      z_bf, E1, 0, o_w, E, 0, d_out, E, 0, R, E, E1,
      o_b, nullptr, nullptr, nullptr, nullptr, L, Nb);
}
